// SpikeAttention_47622597378556
// MI455X (gfx1250) — matrix operand order repaired and checked by reading, not compiled, not run on hardware
//
#include <hip/hip_runtime.h>
#include <math.h>

// ---------------- CDNA5 WMMA plumbing (gfx1250, wave32) ----------------
typedef __attribute__((ext_vector_type(16))) __bf16 v16bf;
typedef __attribute__((ext_vector_type(8)))  float  v8f;

union FragU { uint4 q[2]; v16bf v; };

__device__ __forceinline__ unsigned short f32_to_bf16(float f) {
    unsigned int u = __float_as_uint(f);
    u += 0x7FFFu + ((u >> 16) & 1u);          // round-to-nearest-even
    return (unsigned short)(u >> 16);
}
__device__ __forceinline__ float bf16_to_f32(unsigned short u) {
    return __uint_as_float(((unsigned int)u) << 16);
}

// A fragment: 16x32 bf16 tile, A row-major, base = &A[row0*lda + k0]
// lanes 0-15 hold row M=l, K in {0..7, 16..23}; lanes 16-31: K in {8..15, 24..31}
__device__ __forceinline__ v16bf load_a_frag(const unsigned short* base, int lda, int lane) {
    int r  = lane & 15;
    int kh = lane >> 4;
    FragU f;
    const unsigned short* p = base + (size_t)r * lda + kh * 8;
    f.q[0] = *(const uint4*)(p);
    f.q[1] = *(const uint4*)(p + 16);
    return f.v;
}

// B fragment: 32x16 bf16 tile of W^T, W row-major [outCol][K], base = &W[col0*ldw + k0]
// lanes 0-15 hold col N=l, K in {0..7, 16..23}; lanes 16-31: K in {8..15, 24..31}
__device__ __forceinline__ v16bf load_b_frag(const unsigned short* base, int ldw, int lane) {
    int c  = lane & 15;
    int kg = lane >> 4;
    FragU f;
    const unsigned short* p = base + (size_t)c * ldw + kg * 8;
    f.q[0] = *(const uint4*)(p);
    f.q[1] = *(const uint4*)(p + 16);
    return f.v;
}

__device__ __forceinline__ v8f wmma_bf16(v16bf a, v16bf b, v8f c) {
    return __builtin_amdgcn_wmma_f32_16x16x32_bf16(
        false, a, false, b, (short)0, c, false, false);
}

// Store a 16x16 bf16 tile row-major via per-wave LDS transpose (coalesced 16B/lane).
// tmp[r] = D[8*(lane>>4)+r][lane&15]. dst points at tile origin. Uniform across block.
__device__ __forceinline__ void tile_store_bf16_rm(unsigned short* dst, size_t ldd,
                                                   const unsigned short (&tmp)[8],
                                                   int lane, unsigned short* lbuf) {
    int cl = lane & 15, hl = lane >> 4;
    __syncthreads();
#pragma unroll
    for (int r = 0; r < 8; ++r) lbuf[(hl * 8 + r) * 16 + cl] = tmp[r];
    __syncthreads();
    int row = lane >> 1, half = lane & 1;
    uint4 v = *(const uint4*)(lbuf + row * 16 + half * 8);
    *(uint4*)(dst + (size_t)row * ldd + half * 8) = v;
}

// ---------------- problem constants ----------------
#define BATCH 4
#define NTOK  4096
#define DIM   1024
#define HEADS 16
#define HDIM  64
#define TOKS  (BATCH * NTOK)          // 16384

// ---------------- kernels ----------------

// f32 -> bf16, 4-wide
__global__ void cvt_kernel(const float* __restrict__ in, unsigned short* __restrict__ out, int n4) {
    int i = blockIdx.x * blockDim.x + threadIdx.x;
    if (i < n4) {
        float4 f = ((const float4*)in)[i];
        union { unsigned short s[4]; uint2 q; } u;
        u.s[0] = f32_to_bf16(f.x); u.s[1] = f32_to_bf16(f.y);
        u.s[2] = f32_to_bf16(f.z); u.s[3] = f32_to_bf16(f.w);
        ((uint2*)out)[i] = u.q;
    }
}

// Projection GEMM: Out = epi(A @ W^T + bias). A: (TOKS, DIM) bf16, W: (DIM, DIM) bf16.
// MODE 0: elu+1, row-major bf16 (Q).  MODE 1: elu+1, transposed (B,DIM,NTOK) (Kt).
// MODE 2: identity, transposed (Vt).
// Block 256 thr = 8 waves (2M x 4N), block tile 128x128, wave tile 64x32.
template <int MODE>
__global__ __launch_bounds__(256) void gemm_proj(const unsigned short* __restrict__ A,
                                                 const unsigned short* __restrict__ W,
                                                 const float* __restrict__ bias,
                                                 unsigned short* __restrict__ Out) {
    __shared__ unsigned short tb[8][256];
    int lane = threadIdx.x & 31, wave = threadIdx.x >> 5;
    int wM = wave >> 2, wN = wave & 3;
    int mbase = blockIdx.y * 128 + wM * 64;
    int nbase = blockIdx.x * 128 + wN * 32;

    v8f acc[4][2] = {};
    for (int k0 = 0; k0 < DIM; k0 += 32) {
        v16bf b0 = load_b_frag(W + (size_t)nbase * DIM + k0, DIM, lane);
        v16bf b1 = load_b_frag(W + (size_t)(nbase + 16) * DIM + k0, DIM, lane);
#pragma unroll
        for (int i = 0; i < 4; ++i) {
            v16bf a = load_a_frag(A + (size_t)(mbase + i * 16) * DIM + k0, DIM, lane);
            acc[i][0] = wmma_bf16(a, b0, acc[i][0]);
            acc[i][1] = wmma_bf16(a, b1, acc[i][1]);
        }
    }

    int cl = lane & 15, hl = lane >> 4;
#pragma unroll
    for (int i = 0; i < 4; ++i) {
#pragma unroll
        for (int j = 0; j < 2; ++j) {
            int col  = nbase + j * 16 + cl;
            float bv = bias[col];
            int row0 = mbase + i * 16 + hl * 8;
            unsigned short tmp[8];
#pragma unroll
            for (int r = 0; r < 8; ++r) {
                float v = acc[i][j][r] + bv;
                if (MODE != 2) v = (v > 0.f) ? (v + 1.f) : __expf(v);   // elu(v)+1
                tmp[r] = f32_to_bf16(v);
            }
            if (MODE == 0) {
                unsigned short* dst = Out + (size_t)(mbase + i * 16) * DIM + (nbase + j * 16);
                tile_store_bf16_rm(dst, DIM, tmp, lane, &tb[wave][0]);
            } else {
                // transposed store: tokens contiguous -> one b128 store per lane
                int bb = row0 >> 12, n = row0 & (NTOK - 1);
                union { unsigned short s[8]; uint4 q; } u;
#pragma unroll
                for (int r = 0; r < 8; ++r) u.s[r] = tmp[r];
                *(uint4*)(Out + (size_t)bb * (DIM * NTOK) + (size_t)col * NTOK + n) = u.q;
            }
        }
    }
}

// Per-(b,h): KVt[v][d] = sum_n Vt[v][n]*Kt[d][n]  (64x64, split-K over 8 waves,
// ds_add_f32 reduction in LDS) and Ksum[d] = sum_n Kt[d][n].
__global__ __launch_bounds__(256) void kv_head_kernel(const unsigned short* __restrict__ Kt,
                                                      const unsigned short* __restrict__ Vt,
                                                      unsigned short* __restrict__ KVt,
                                                      float* __restrict__ Ksum) {
    __shared__ float kv[64 * 64];
    __shared__ float ks[64];
    int bh   = blockIdx.x;                      // b*HEADS + h
    size_t base = (size_t)(bh >> 4) * (DIM * NTOK) + (size_t)((bh & 15) * HDIM) * NTOK;
    const unsigned short* Kh = Kt + base;       // rows d, ld NTOK
    const unsigned short* Vh = Vt + base;       // rows v, ld NTOK
    int tid = threadIdx.x, lane = tid & 31, wave = tid >> 5;

    for (int i = tid; i < 64 * 64; i += 256) kv[i] = 0.f;
    if (tid < 64) ks[tid] = 0.f;
    __syncthreads();

    v8f acc[4][4] = {};
    int kbeg = wave * (NTOK / 8);
    for (int k0 = kbeg; k0 < kbeg + NTOK / 8; k0 += 32) {
        v16bf a[4], bfr[4];
#pragma unroll
        for (int i = 0; i < 4; ++i) a[i]   = load_a_frag(Vh + (size_t)(i * 16) * NTOK + k0, NTOK, lane);
#pragma unroll
        for (int j = 0; j < 4; ++j) bfr[j] = load_b_frag(Kh + (size_t)(j * 16) * NTOK + k0, NTOK, lane);
#pragma unroll
        for (int i = 0; i < 4; ++i)
#pragma unroll
            for (int j = 0; j < 4; ++j)
                acc[i][j] = wmma_bf16(a[i], bfr[j], acc[i][j]);
    }

    int cl = lane & 15, hl = lane >> 4;
#pragma unroll
    for (int i = 0; i < 4; ++i)
#pragma unroll
        for (int j = 0; j < 4; ++j)
#pragma unroll
            for (int r = 0; r < 8; ++r)
                atomicAdd(&kv[(i * 16 + hl * 8 + r) * 64 + (j * 16 + cl)], acc[i][j][r]);

    // K_sum: thread -> (d = tid&63, chunk = tid>>6 of 1024 tokens)
    {
        int d = tid & 63, chunk = tid >> 6;
        const unsigned short* p = Kh + (size_t)d * NTOK + chunk * 1024;
        float s = 0.f;
        for (int t = 0; t < 1024; t += 8) {
            union { uint4 q; unsigned short s[8]; } u;
            u.q = *(const uint4*)(p + t);
#pragma unroll
            for (int e = 0; e < 8; ++e) s += bf16_to_f32(u.s[e]);
        }
        atomicAdd(&ks[d], s);
    }
    __syncthreads();

    {   // store KVt (bf16) — 16 consecutive elements per thread
        int idx0 = tid * 16;
        union { unsigned short s[16]; uint4 q[2]; } u;
#pragma unroll
        for (int e = 0; e < 16; ++e) u.s[e] = f32_to_bf16(kv[idx0 + e]);
        unsigned short* dst = KVt + (size_t)bh * 4096 + idx0;
        ((uint4*)dst)[0] = u.q[0];
        ((uint4*)dst)[1] = u.q[1];
    }
    if (tid < 64) Ksum[bh * 64 + tid] = ks[tid];
}

// denom[b,h,n] = Q[b,n,h*64..] . Ksum[b,h,:] + 1e-6
__global__ void denom_kernel(const unsigned short* __restrict__ Qb,
                             const float* __restrict__ Ksum,
                             float* __restrict__ denom) {
    int i = blockIdx.x * 256 + threadIdx.x;     // b*65536 + h*4096 + n
    int b = i >> 16, h = (i >> 12) & 15, n = i & (NTOK - 1);
    const unsigned short* q = Qb + (size_t)(b * NTOK + n) * DIM + h * HDIM;
    const float* ks = Ksum + (b * HEADS + h) * HDIM;
    float s = 0.f;
#pragma unroll 8
    for (int d = 0; d < HDIM; ++d) s += bf16_to_f32(q[d]) * ks[d];
    denom[i] = s + 1e-6f;
}

// attn = (Q @ KVt^T) / denom * sigmoid(stdp[h]); store bf16 row-major (b*n, DIM)
// grid (NTOK/128, B*HEADS); 8 waves = 2M x 4N; wave tile 64x16, K=64 (2 wmma steps)
__global__ __launch_bounds__(256) void attn_kernel(const unsigned short* __restrict__ Qb,
                                                   const unsigned short* __restrict__ KVt,
                                                   const float* __restrict__ denom,
                                                   const float* __restrict__ stdp,
                                                   unsigned short* __restrict__ attnb) {
    __shared__ unsigned short tb[8][256];
    int bh = blockIdx.y, b = bh >> 4, h = bh & 15;
    int lane = threadIdx.x & 31, wave = threadIdx.x >> 5;
    int wM = wave >> 2, wN = wave & 3;
    int mbase = blockIdx.x * 128 + wM * 64;     // token within batch
    int nb = wN * 16;                           // v column within head
    const unsigned short* Qh = Qb + (size_t)(b * NTOK) * DIM + h * HDIM;
    const unsigned short* Wh = KVt + (size_t)bh * 4096;    // rows v, ld 64
    float scale = 1.f / (1.f + __expf(-stdp[h]));

    v8f acc[4] = {};
#pragma unroll
    for (int k0 = 0; k0 < HDIM; k0 += 32) {
        v16bf bfr = load_b_frag(Wh + nb * HDIM + k0, HDIM, lane);
#pragma unroll
        for (int i = 0; i < 4; ++i) {
            v16bf a = load_a_frag(Qh + (size_t)(mbase + i * 16) * DIM + k0, DIM, lane);
            acc[i] = wmma_bf16(a, bfr, acc[i]);
        }
    }

    int hl = lane >> 4;
    const float* dh = denom + (size_t)bh * NTOK;
#pragma unroll
    for (int i = 0; i < 4; ++i) {
        int r0 = mbase + i * 16 + hl * 8;
        unsigned short tmp[8];
#pragma unroll
        for (int r = 0; r < 8; ++r)
            tmp[r] = f32_to_bf16(acc[i][r] * scale / dh[r0 + r]);
        unsigned short* dst = attnb + (size_t)(b * NTOK + mbase + i * 16) * DIM + (h * HDIM + nb);
        tile_store_bf16_rm(dst, DIM, tmp, lane, &tb[wave][0]);
    }
}

// Final: out = attn @ Wo^T + bo, f32 to d_out
__global__ __launch_bounds__(256) void gemm_out(const unsigned short* __restrict__ A,
                                                const unsigned short* __restrict__ W,
                                                const float* __restrict__ bias,
                                                float* __restrict__ Out) {
    int lane = threadIdx.x & 31, wave = threadIdx.x >> 5;
    int wM = wave >> 2, wN = wave & 3;
    int mbase = blockIdx.y * 128 + wM * 64;
    int nbase = blockIdx.x * 128 + wN * 32;

    v8f acc[4][2] = {};
    for (int k0 = 0; k0 < DIM; k0 += 32) {
        v16bf b0 = load_b_frag(W + (size_t)nbase * DIM + k0, DIM, lane);
        v16bf b1 = load_b_frag(W + (size_t)(nbase + 16) * DIM + k0, DIM, lane);
#pragma unroll
        for (int i = 0; i < 4; ++i) {
            v16bf a = load_a_frag(A + (size_t)(mbase + i * 16) * DIM + k0, DIM, lane);
            acc[i][0] = wmma_bf16(a, b0, acc[i][0]);
            acc[i][1] = wmma_bf16(a, b1, acc[i][1]);
        }
    }

    int cl = lane & 15, hl = lane >> 4;
#pragma unroll
    for (int i = 0; i < 4; ++i)
#pragma unroll
        for (int j = 0; j < 2; ++j) {
            int col  = nbase + j * 16 + cl;
            float bv = bias[col];
            int row0 = mbase + i * 16 + hl * 8;
#pragma unroll
            for (int r = 0; r < 8; ++r)
                Out[(size_t)(row0 + r) * DIM + col] = acc[i][j][r] + bv;
        }
}

__global__ void tail_kernel(const float* __restrict__ stdp, float* __restrict__ out) {
    int i = threadIdx.x;
    if (i < HEADS) out[(size_t)TOKS * DIM + i] = stdp[i];
}

// ---------------- host launcher ----------------
extern "C" void kernel_launch(void* const* d_in, const int* in_sizes, int n_in,
                              void* d_out, int out_size, void* d_ws, size_t ws_size,
                              hipStream_t stream) {
    (void)in_sizes; (void)n_in; (void)out_size; (void)ws_size;

    const float* x    = (const float*)d_in[0];
    const float* Wq   = (const float*)d_in[1];
    const float* bq   = (const float*)d_in[2];
    const float* Wk   = (const float*)d_in[3];
    const float* bk   = (const float*)d_in[4];
    const float* Wv   = (const float*)d_in[5];
    const float* bv   = (const float*)d_in[6];
    const float* Wo   = (const float*)d_in[7];
    const float* bo   = (const float*)d_in[8];
    const float* stdp = (const float*)d_in[9];
    float* out = (float*)d_out;

    const size_t MiB = 1ull << 20;
    char* ws = (char*)d_ws;
    unsigned short* xb    = (unsigned short*)(ws + 0 * MiB);    // 32 MiB
    unsigned short* Wqb   = (unsigned short*)(ws + 32 * MiB);   //  2 MiB
    unsigned short* Wkb   = (unsigned short*)(ws + 34 * MiB);
    unsigned short* Wvb   = (unsigned short*)(ws + 36 * MiB);
    unsigned short* Wob   = (unsigned short*)(ws + 38 * MiB);
    unsigned short* Qb    = (unsigned short*)(ws + 40 * MiB);   // 32 MiB (b*n, DIM)
    unsigned short* Kt    = (unsigned short*)(ws + 72 * MiB);   // 32 MiB (B, DIM, NTOK)
    unsigned short* Vt    = (unsigned short*)(ws + 104 * MiB);  // 32 MiB
    unsigned short* attnb = (unsigned short*)(ws + 136 * MiB);  // 32 MiB
    unsigned short* KVt   = (unsigned short*)(ws + 168 * MiB);  // 0.5 MiB (B*H, 64*64)
    float* Ksum           = (float*)(ws + 169 * MiB);           // 16 KiB
    float* denom          = (float*)(ws + 170 * MiB);           //  1 MiB

    // 1) convert to bf16
    cvt_kernel<<<(TOKS * DIM / 4 + 255) / 256, 256, 0, stream>>>(x, xb, TOKS * DIM / 4);
    cvt_kernel<<<(DIM * DIM / 4 + 255) / 256, 256, 0, stream>>>(Wq, Wqb, DIM * DIM / 4);
    cvt_kernel<<<(DIM * DIM / 4 + 255) / 256, 256, 0, stream>>>(Wk, Wkb, DIM * DIM / 4);
    cvt_kernel<<<(DIM * DIM / 4 + 255) / 256, 256, 0, stream>>>(Wv, Wvb, DIM * DIM / 4);
    cvt_kernel<<<(DIM * DIM / 4 + 255) / 256, 256, 0, stream>>>(Wo, Wob, DIM * DIM / 4);

    // 2) projections (WMMA): Q (elu+1, row-major), Kt/Vt (transposed)
    dim3 gproj(DIM / 128, TOKS / 128);
    gemm_proj<0><<<gproj, 256, 0, stream>>>(xb, Wqb, bq, Qb);
    gemm_proj<1><<<gproj, 256, 0, stream>>>(xb, Wkb, bk, Kt);
    gemm_proj<2><<<gproj, 256, 0, stream>>>(xb, Wvb, bv, Vt);

    // 3) per-head KV^T (WMMA) + K_sum
    kv_head_kernel<<<BATCH * HEADS, 256, 0, stream>>>(Kt, Vt, KVt, Ksum);

    // 4) denominator
    denom_kernel<<<BATCH * HEADS * NTOK / 256, 256, 0, stream>>>(Qb, Ksum, denom);

    // 5) attention numerator (WMMA) + normalize + gate
    attn_kernel<<<dim3(NTOK / 128, BATCH * HEADS), 256, 0, stream>>>(Qb, KVt, denom, stdp, attnb);

    // 6) output projection (WMMA, f32 out)
    gemm_out<<<gproj, 256, 0, stream>>>(attnb, Wob, bo, out);

    // 7) append stdp
    tail_kernel<<<1, 32, 0, stream>>>(stdp, out);
}
